// MultiHeadSelfAttention_40089224741526
// MI455X (gfx1250) — matrix operand order repaired and checked by reading, not compiled, not run on hardware
//
#include <hip/hip_runtime.h>

#define B_  32
#define S_  577
#define SP_ 608   // padded seq len: 19 * 32
#define E_  768
#define H_  12
#define D_  64
#define HD_ 768

typedef __attribute__((ext_vector_type(16))) __bf16 v16bf;
typedef __attribute__((ext_vector_type(8)))  float  v8f;
typedef __attribute__((ext_vector_type(4)))  unsigned int u32x4;

struct Frag { u32x4 lo, hi; };

__device__ __forceinline__ v16bf fragv(const Frag& f) {
  return __builtin_bit_cast(v16bf, f);
}

// round-to-nearest-even f32 -> bf16 (finite inputs)
__device__ __forceinline__ unsigned short f2bf(float f) {
  unsigned int u = __builtin_bit_cast(unsigned int, f);
  u += 0x7FFFu + ((u >> 16) & 1u);
  return (unsigned short)(u >> 16);
}

// A fragment 16x32 bf16 (M x K), row-major source, row stride rs (elements).
// lanes 0-15: M=lane, v0..3=K 0..7, v4..7=K 16..23 ; lanes 16-31: +8 on K.
__device__ __forceinline__ Frag load_a(const unsigned short* __restrict__ base,
                                       int row0, int k0, int rs, int lane) {
  int m = lane & 15, hi = lane >> 4;
  const unsigned short* p = base + (size_t)(row0 + m) * rs + k0 + hi * 8;
  Frag f;
  f.lo = *(const u32x4*)(p);        // K = hi*8 .. hi*8+7
  f.hi = *(const u32x4*)(p + 16);   // K = 16+hi*8 .. 16+hi*8+7
  return f;
}

// B fragment 32x16 bf16 (K x N), K-major source with row stride rs (elements).
// lanes 0-15: N=lane, elems 0..7=K 0..7, 8..15=K 16..23 ; lanes 16-31: +8 on K.
__device__ __forceinline__ Frag load_b(const unsigned short* __restrict__ base,
                                       int k0, int n0, int rs, int lane) {
  int n = lane & 15, hi = lane >> 4;
  const unsigned short* p = base + (size_t)(k0 + hi * 8) * rs + n0 + n;
  unsigned short t[16];
#pragma unroll
  for (int i = 0; i < 8; i++) {
    t[i] = p[(size_t)i * rs];
    t[8 + i] = p[(size_t)(16 + i) * rs];
  }
  Frag f = __builtin_bit_cast(Frag, t);
  return f;
}

__device__ __forceinline__ v8f wmma_bf16(const Frag& a, const Frag& b, v8f c) {
  return __builtin_amdgcn_wmma_f32_16x16x32_bf16(false, fragv(a), false, fragv(b),
                                                 (short)0, c, false, false);
}

__device__ __forceinline__ float lane_xor(float v, int mask) {
  int lane = (int)(threadIdx.x & 31);
  int j = __builtin_amdgcn_ds_bpermute(((lane ^ mask) << 2),
                                       __builtin_bit_cast(int, v));
  return __builtin_bit_cast(float, j);
}

// ---------------- prep kernels ----------------

__global__ void k_cast_x(const float* __restrict__ x,
                         unsigned short* __restrict__ xb, int total) {
  int i = blockIdx.x * blockDim.x + threadIdx.x;
  if (i >= total) return;
  int e = i % E_;
  int s = (i / E_) % SP_;
  int b = i / (E_ * SP_);
  float v = (s < S_) ? x[((size_t)b * S_ + s) * E_ + e] : 0.0f;
  xb[i] = f2bf(v);
}

// wT[m][h][e][d] = w_m[h][d][e]  (bf16), m in {q,k,v}
__global__ void k_prep_w(const float* __restrict__ wq, const float* __restrict__ wk,
                         const float* __restrict__ wv,
                         unsigned short* __restrict__ wT, int total) {
  int i = blockIdx.x * blockDim.x + threadIdx.x;
  if (i >= total) return;                      // total = 3*H*E*D
  int d = i % D_;
  int e = (i / D_) % E_;
  int h = (i / (D_ * E_)) % H_;
  int m = i / (D_ * E_ * H_);
  const float* w = (m == 0) ? wq : (m == 1) ? wk : wv;
  wT[i] = f2bf(w[((size_t)h * D_ + d) * E_ + e]);
}

// wpT[k][n] = w_proj[n][k]  (bf16)
__global__ void k_prep_wp(const float* __restrict__ wp,
                          unsigned short* __restrict__ wpT, int total) {
  int i = blockIdx.x * blockDim.x + threadIdx.x;
  if (i >= total) return;                      // total = HD*E
  int n = i % E_;
  int k = i / E_;
  wpT[i] = f2bf(wp[(size_t)n * HD_ + k]);
}

// ---------------- QKV projection: one wave -> 16x64 tile ----------------
// grid: (SP/16, B*H, 3). z selects q/k/v.  K stored transposed [D, SP].
__global__ __launch_bounds__(32) void k_qkv(const unsigned short* __restrict__ xb,
    const unsigned short* __restrict__ wT,
    unsigned short* __restrict__ Q, unsigned short* __restrict__ Kt,
    unsigned short* __restrict__ V) {
  int lane = threadIdx.x;
  int m = lane & 15, hi = lane >> 4;
  int sblk = blockIdx.x;
  int bh = blockIdx.y;
  int h = bh % H_;
  int which = blockIdx.z;
  const unsigned short* xrow = xb + (size_t)(bh / H_) * SP_ * E_;        // [SP,E]
  const unsigned short* wb = wT + ((size_t)which * H_ + h) * (E_ * D_);  // [E,D]

  v8f acc[4] = {};
  for (int k0 = 0; k0 < E_; k0 += 32) {
    Frag a = load_a(xrow, sblk * 16, k0, E_, lane);
#pragma unroll
    for (int t = 0; t < 4; t++) {
      Frag b = load_b(wb, k0, t * 16, D_, lane);
      acc[t] = wmma_bf16(a, b, acc[t]);
    }
  }

  if (which == 1) {
    // Kt[b,h,d,s]: per lane d fixed (N), s contiguous over regs -> packed b128
    unsigned short* dst = Kt + (size_t)bh * (D_ * SP_);
#pragma unroll
    for (int t = 0; t < 4; t++) {
      int d = t * 16 + m;
      u32x4 pk = {
        (unsigned int)f2bf(acc[t][0]) | ((unsigned int)f2bf(acc[t][1]) << 16),
        (unsigned int)f2bf(acc[t][2]) | ((unsigned int)f2bf(acc[t][3]) << 16),
        (unsigned int)f2bf(acc[t][4]) | ((unsigned int)f2bf(acc[t][5]) << 16),
        (unsigned int)f2bf(acc[t][6]) | ((unsigned int)f2bf(acc[t][7]) << 16)};
      *(u32x4*)(dst + (size_t)d * SP_ + sblk * 16 + hi * 8) = pk;
    }
  } else {
    unsigned short* dst = (which == 0 ? Q : V) + (size_t)bh * (SP_ * D_);
#pragma unroll
    for (int t = 0; t < 4; t++)
#pragma unroll
      for (int r = 0; r < 8; r++)
        dst[(size_t)(sblk * 16 + r + 8 * hi) * D_ + t * 16 + m] = f2bf(acc[t][r]);
  }
}

// ---------------- fused flash attention: one wave -> 16 query rows ----------------
// grid: (SP/16, B*H)
__global__ __launch_bounds__(32) void k_attn(const unsigned short* __restrict__ Q,
    const unsigned short* __restrict__ Kt, const unsigned short* __restrict__ V,
    unsigned short* __restrict__ cat) {
  __shared__ unsigned short Pl[16 * 32];
  int lane = threadIdx.x;
  int m = lane & 15, hi = lane >> 4;
  int sblk = blockIdx.x;
  int bh = blockIdx.y;
  int b = bh / H_, h = bh % H_;
  const unsigned short* Qb  = Q  + (size_t)bh * SP_ * D_;  // [SP,D]
  const unsigned short* Ktb = Kt + (size_t)bh * D_ * SP_;  // [D,SP]
  const unsigned short* Vb  = V  + (size_t)bh * SP_ * D_;  // [SP,D]

  Frag aq0 = load_a(Qb, sblk * 16, 0,  D_, lane);
  Frag aq1 = load_a(Qb, sblk * 16, 32, D_, lane);

  v8f o[4] = {};
  float mrow[8], lrow[8];
#pragma unroll
  for (int r = 0; r < 8; r++) { mrow[r] = -3.0e38f; lrow[r] = 0.f; }
  const float scale = 0.125f;  // 1/sqrt(64)

  for (int tb = 0; tb < SP_; tb += 32) {
    v8f sc[2];
#pragma unroll
    for (int sub = 0; sub < 2; sub++) {
      v8f s = {};
      Frag b0 = load_b(Ktb, 0,  tb + sub * 16, SP_, lane);
      s = wmma_bf16(aq0, b0, s);
      Frag b1 = load_b(Ktb, 32, tb + sub * 16, SP_, lane);
      s = wmma_bf16(aq1, b1, s);
      int tcol = tb + sub * 16 + m;                 // lane's key column
      float msk = (tcol < S_) ? 0.0f : -3.0e38f;    // mask padded keys
#pragma unroll
      for (int r = 0; r < 8; r++) s[r] = s[r] * scale + msk;
      sc[sub] = s;
    }
    // online softmax: row max/sum across the 16 lanes of each half
    float alpha[8];
#pragma unroll
    for (int r = 0; r < 8; r++) {
      float v = fmaxf(sc[0][r], sc[1][r]);
      v = fmaxf(v, lane_xor(v, 1));
      v = fmaxf(v, lane_xor(v, 2));
      v = fmaxf(v, lane_xor(v, 4));
      v = fmaxf(v, lane_xor(v, 8));
      float mn = fmaxf(mrow[r], v);
      alpha[r] = __expf(mrow[r] - mn);
      mrow[r] = mn;
    }
#pragma unroll
    for (int sub = 0; sub < 2; sub++)
#pragma unroll
      for (int r = 0; r < 8; r++) sc[sub][r] = __expf(sc[sub][r] - mrow[r]);
#pragma unroll
    for (int r = 0; r < 8; r++) {
      float v = sc[0][r] + sc[1][r];
      v += lane_xor(v, 1);
      v += lane_xor(v, 2);
      v += lane_xor(v, 4);
      v += lane_xor(v, 8);
      lrow[r] = lrow[r] * alpha[r] + v;
    }
#pragma unroll
    for (int t = 0; t < 4; t++)
#pragma unroll
      for (int r = 0; r < 8; r++) o[t][r] *= alpha[r];

    // C-layout -> A-layout relayout of P through LDS (16x32 bf16, row major)
    __syncthreads();
#pragma unroll
    for (int sub = 0; sub < 2; sub++)
#pragma unroll
      for (int r = 0; r < 8; r++)
        Pl[(r + 8 * hi) * 32 + sub * 16 + m] = f2bf(sc[sub][r]);
    __syncthreads();
    Frag pa;
    {
      const unsigned short* pp = &Pl[m * 32 + hi * 8];
      pa.lo = *(const u32x4*)(pp);
      pa.hi = *(const u32x4*)(pp + 16);
    }
#pragma unroll
    for (int t = 0; t < 4; t++) {
      Frag vb = load_b(Vb, tb, t * 16, D_, lane);
      o[t] = wmma_bf16(pa, vb, o[t]);
    }
  }

  // normalize and store concat [B, SP, H*D] in bf16
  unsigned short* dst = cat + (size_t)b * SP_ * HD_ + (size_t)h * D_;
#pragma unroll
  for (int t = 0; t < 4; t++)
#pragma unroll
    for (int r = 0; r < 8; r++) {
      int s = sblk * 16 + r + 8 * hi;
      dst[(size_t)s * HD_ + t * 16 + m] = f2bf(o[t][r] / lrow[r]);
    }
}

// ---------------- output projection + bias: one wave -> 16x64 tile ----------------
// grid: (SP/16, B, HD/64)
__global__ __launch_bounds__(32) void k_proj(const unsigned short* __restrict__ cat,
    const unsigned short* __restrict__ wpT, const float* __restrict__ bias,
    float* __restrict__ out) {
  int lane = threadIdx.x;
  int m = lane & 15, hi = lane >> 4;
  int sblk = blockIdx.x;
  int b = blockIdx.y;
  int nblk = blockIdx.z;
  const unsigned short* arow = cat + (size_t)b * SP_ * HD_;
  v8f acc[4] = {};
  for (int k0 = 0; k0 < HD_; k0 += 32) {
    Frag a = load_a(arow, sblk * 16, k0, HD_, lane);
#pragma unroll
    for (int t = 0; t < 4; t++) {
      Frag w = load_b(wpT, k0, nblk * 64 + t * 16, E_, lane);
      acc[t] = wmma_bf16(a, w, acc[t]);
    }
  }
#pragma unroll
  for (int t = 0; t < 4; t++) {
    int n = nblk * 64 + t * 16 + m;
    float bv = bias[n];
#pragma unroll
    for (int r = 0; r < 8; r++) {
      int s = sblk * 16 + r + 8 * hi;
      if (s < S_) out[((size_t)b * S_ + s) * E_ + n] = acc[t][r] + bv;
    }
  }
}

extern "C" void kernel_launch(void* const* d_in, const int* in_sizes, int n_in,
                              void* d_out, int out_size, void* d_ws, size_t ws_size,
                              hipStream_t stream) {
  const float* x  = (const float*)d_in[0];
  const float* wq = (const float*)d_in[1];
  const float* wk = (const float*)d_in[2];
  const float* wv = (const float*)d_in[3];
  const float* wp = (const float*)d_in[4];
  const float* bp = (const float*)d_in[5];
  float* out = (float*)d_out;

  char* p = (char*)d_ws;
  const size_t xb_elems  = (size_t)B_ * SP_ * E_;       // also cat size
  const size_t wT_elems  = (size_t)3 * H_ * E_ * D_;
  const size_t wp_elems  = (size_t)HD_ * E_;
  const size_t qkv_elems = (size_t)B_ * H_ * SP_ * D_;

  unsigned short* xb  = (unsigned short*)p; p += xb_elems  * 2;
  unsigned short* wT  = (unsigned short*)p; p += wT_elems  * 2;
  unsigned short* wpT = (unsigned short*)p; p += wp_elems  * 2;
  unsigned short* Q   = (unsigned short*)p; p += qkv_elems * 2;
  unsigned short* Kt  = (unsigned short*)p; p += qkv_elems * 2;
  unsigned short* V   = (unsigned short*)p; p += qkv_elems * 2;
  unsigned short* cat = (unsigned short*)p; p += xb_elems  * 2;

  k_cast_x<<<(unsigned)((xb_elems + 255) / 256), 256, 0, stream>>>(x, xb, (int)xb_elems);
  k_prep_w<<<(unsigned)((wT_elems + 255) / 256), 256, 0, stream>>>(wq, wk, wv, wT, (int)wT_elems);
  k_prep_wp<<<(unsigned)((wp_elems + 255) / 256), 256, 0, stream>>>(wp, wpT, (int)wp_elems);

  dim3 g2(SP_ / 16, B_ * H_, 3);
  k_qkv<<<g2, 32, 0, stream>>>(xb, wT, Q, Kt, V);

  dim3 g3(SP_ / 16, B_ * H_);
  k_attn<<<g3, 32, 0, stream>>>(Q, Kt, V, cat);

  dim3 g4(SP_ / 16, B_, HD_ / 64);
  k_proj<<<g4, 32, 0, stream>>>(cat, wpT, bp, out);
}
